// realNVP_78993038508074
// MI455X (gfx1250) — matrix operand order repaired and checked by reading, not compiled, not run on hardware
//
#include <hip/hip_runtime.h>
#include <hip/hip_bf16.h>
#include <math.h>

// ---------------------------------------------------------------------------
// RealNVP flow, MI455X (gfx1250, wave32, WMMA).
//
// The only large tensor that hits HBM is the pre-BN hidden activation h2pre
// (B x 128, f16, 64MB), written once and read once per layer. h1 is generated
// in registers inside the WMMA GEMM (first linear has input dim 1; BN1 stats
// follow from scalar stats of xin; biases b1/b2 cancel under train-mode BN).
// ~1.1 GB total HBM traffic -> ~46us at 23.3 TB/s; 69 GFLOP of
// v_wmma_f32_16x16x32_f16 is negligible.
//
// Round-2 change: per-lane A coefficients are pre-packed as half2 register
// arrays (64 VGPRs instead of ~128 f32) so the GEMM loop no longer spills to
// scratch (round 1 showed "Folded Reload" scratch traffic in the hot loop).
// ---------------------------------------------------------------------------

#define HH 128
#define BB 262144
#define LL 8
#define BN_EPS   1e-5f
#define FLOW_EPS 1e-4f
#define GEMM_BLOCKS 2048

typedef __attribute__((ext_vector_type(16))) _Float16 v16h;
typedef __attribute__((ext_vector_type(8)))  _Float16 v8h;
typedef __attribute__((ext_vector_type(2)))  _Float16 v2h;
typedef __attribute__((ext_vector_type(8)))  float    v8f;

// ---- init: copy x into mutable state, zero sldj ---------------------------
__global__ void k_init(const float* __restrict__ x, float* xcur, float* sldj) {
  int i = blockIdx.x * blockDim.x + threadIdx.x;
  if (i < BB) {
    xcur[2 * i]     = x[2 * i];
    xcur[2 * i + 1] = x[2 * i + 1];
    sldj[i] = 0.f;
  }
}

__global__ void k_zero(float* p, int n) {
  int i = blockIdx.x * blockDim.x + threadIdx.x;
  if (i < n) p[i] = 0.f;
}

// ---- batch stats of xin (scalar column) -----------------------------------
__global__ void k_ustat(const float* __restrict__ xcur, float* ustats, int xin_col) {
  __shared__ float ss[256], sq[256];
  float s = 0.f, q = 0.f;
  for (int i = blockIdx.x * blockDim.x + threadIdx.x; i < BB;
       i += gridDim.x * blockDim.x) {
    float u = xcur[2 * i + xin_col];
    s += u; q += u * u;
  }
  ss[threadIdx.x] = s; sq[threadIdx.x] = q;
  __syncthreads();
  for (int o = 128; o > 0; o >>= 1) {
    if (threadIdx.x < o) {
      ss[threadIdx.x] += ss[threadIdx.x + o];
      sq[threadIdx.x] += sq[threadIdx.x + o];
    }
    __syncthreads();
  }
  if (threadIdx.x == 0) {
    atomicAdd(&ustats[0], ss[0]);
    atomicAdd(&ustats[1], sq[0]);
  }
}

// ---- per-layer prep: fold WN+BN1 into (alpha,beta); W2 -> f16 -------------
// h1[b,k] = relu(alpha_k * u_b + beta_k). b1 cancels under BN.
__global__ void k_prep(const float* __restrict__ v1, const float* __restrict__ g1,
                       const float* __restrict__ bn1g, const float* __restrict__ bn1b,
                       const float* __restrict__ v2, const float* __restrict__ g2,
                       const float* __restrict__ ustats,
                       float* alpha, float* beta, _Float16* w2h, int l) {
  int j = threadIdx.x;                       // 128 threads
  float mu  = ustats[0] * (1.f / BB);
  float var = ustats[1] * (1.f / BB) - mu * mu;
  float v   = v1[l * HH + j];                // (L,H,1): v/||v|| = sign(v)
  float w1  = g1[l * HH + j] * (v >= 0.f ? 1.f : -1.f);
  float inv = rsqrtf(w1 * w1 * var + BN_EPS);
  float a   = w1 * inv * bn1g[l * HH + j];
  alpha[j]  = a;
  beta[j]   = bn1b[l * HH + j] - a * mu;

  // weight-norm row scaling of W2, emit f16 (b2 cancels under BN2)
  const float* vr = v2 + (size_t)(l * HH + j) * HH;
  float nrm = 0.f;
  for (int k = 0; k < HH; k++) { float t = vr[k]; nrm += t * t; }
  float sc = g2[l * HH + j] * rsqrtf(nrm);
  for (int k = 0; k < HH; k++) w2h[j * HH + k] = (_Float16)(vr[k] * sc);
}

// ---- fused GEMM: h2pre = relu_bn1(u) @ W2^T, + per-feature batch stats ----
// Block: 256 threads = 8 waves. Wave w owns output features [16w,16w+16).
// Grid-stride over 16-row batch tiles. Loop-invariant registers:
//   * B fragments, 4 chunks x v16h                      (32 VGPRs)
//   * A coefficients packed as half2 (alpha,beta) pairs (64 VGPRs)
// Inner loop: 1 f32 load (u) -> 32x v_pk_fma_f16 -> 4x wmma -> stats+store.
__global__ void k_gemm(const float* __restrict__ xcur, const _Float16* __restrict__ w2h,
                       const float* __restrict__ alpha, const float* __restrict__ beta,
                       _Float16* __restrict__ H2, float* h2sum, float* h2sumsq,
                       int xin_col) {
  __shared__ float aS[HH], bS[HH];
  int tid = threadIdx.x;
  if (tid < HH) { aS[tid] = alpha[tid]; bS[tid] = beta[tid]; }
  __syncthreads();

  int wave = tid >> 5, n = tid & 31;
  int nmod = n & 15;
  int col  = wave * 16 + nmod;               // output feature (N) for this lane

  // B (32x16 f16) in contiguous K order: VGPR r holds K=2r,2r+1; lanes>=16 -> K+16.
  v16h bfrag[4];
  #pragma unroll
  for (int c = 0; c < 4; c++) {
    int kb = 32 * c + ((n & 16) ? 16 : 0);
    #pragma unroll
    for (int r = 0; r < 8; r++) {
      v2h p = *(const v2h*)(w2h + col * HH + kb + 2 * r);
      bfrag[c][2 * r] = p[0]; bfrag[c][2 * r + 1] = p[1];
    }
  }

  // A (16x32 f16), same K order as B: VGPR r holds K=2r,2r+1; lanes>=16 -> K+16.
  // Pre-pack this lane's (alpha,beta) in fragment order as half2.
  v2h ah[32], bh[32];
  #pragma unroll
  for (int c = 0; c < 4; c++) {
    int kbase = 32 * c + ((n & 16) ? 16 : 0);
    #pragma unroll
    for (int r = 0; r < 8; r++) {
      int K = kbase + 2 * r;
      v2h av, bv;
      av[0] = (_Float16)aS[K]; av[1] = (_Float16)aS[K + 1];
      bv[0] = (_Float16)bS[K]; bv[1] = (_Float16)bS[K + 1];
      ah[c * 8 + r] = av; bh[c * 8 + r] = bv;
    }
  }

  const _Float16 zh = (_Float16)0;
  float regS = 0.f, regQ = 0.f;
  for (int t = blockIdx.x; t < BB / 16; t += GEMM_BLOCKS) {
    int row0 = t * 16;
    float u = xcur[2 * (row0 + nmod) + xin_col];   // A lane holds row M=n%16
    v2h uh; uh[0] = (_Float16)u; uh[1] = (_Float16)u;
    v8f acc = {};
    #pragma unroll
    for (int c = 0; c < 4; c++) {
      v16h a;
      #pragma unroll
      for (int r = 0; r < 8; r++) {
        v2h tv = ah[c * 8 + r] * uh + bh[c * 8 + r];   // v_pk_fma_f16
        a[2 * r]     = tv[0] > zh ? tv[0] : zh;
        a[2 * r + 1] = tv[1] > zh ? tv[1] : zh;
      }
      acc = __builtin_amdgcn_wmma_f32_16x16x32_f16(
          false, a, false, bfrag[c], (short)0, acc, false, false);
    }
    // Per-feature stats: lane n and lane n^16 share N=col and split the 16 M
    // rows 8/8 (C/D layout), so one shfl_xor(16) gives the full column sum.
    float s8 = 0.f, q8 = 0.f;
    #pragma unroll
    for (int r = 0; r < 8; r++) { float v = acc[r]; s8 += v; q8 += v * v; }
    regS += s8 + __shfl_xor(s8, 16);
    regQ += q8 + __shfl_xor(q8, 16);

    int rbase = row0 + ((n & 16) ? 8 : 0);
    #pragma unroll
    for (int r = 0; r < 8; r++)
      H2[(size_t)(rbase + r) * HH + col] = (_Float16)acc[r];
  }
  if (n < 16) {
    atomicAdd(&h2sum[col],   regS);
    atomicAdd(&h2sumsq[col], regQ);
  }
}

// ---- fold BN2 stats into per-feature affine -------------------------------
__global__ void k_bn2prep(const float* __restrict__ bn2g, const float* __restrict__ bn2b,
                          const float* __restrict__ h2sum, const float* __restrict__ h2sumsq,
                          float* scale2, float* shift2, int l) {
  int j = threadIdx.x;
  float m   = h2sum[j] * (1.f / BB);
  float var = h2sumsq[j] * (1.f / BB) - m * m;
  float sc  = bn2g[l * HH + j] * rsqrtf(var + BN_EPS);
  scale2[j] = sc;
  shift2[j] = bn2b[l * HH + j] - m * sc;
}

// ---- BN2+ReLU+final projection+coupling, 4 lanes per sample ---------------
__global__ void k_out(const _Float16* __restrict__ H2, const float* __restrict__ scale2,
                      const float* __restrict__ shift2, const float* __restrict__ wf,
                      const float* __restrict__ bfv, float* xcur, float* sldj,
                      int l, int xo_col) {
  int gid  = blockIdx.x * blockDim.x + threadIdx.x;
  int samp = gid >> 2, sub = gid & 3;                 // quad covers 256B row
  const _Float16* row = H2 + (size_t)samp * HH + sub * 32;
  const float* w0 = wf + (l * 2 + 0) * HH;
  const float* w1 = wf + (l * 2 + 1) * HH;
  float d0 = 0.f, d1 = 0.f;
  #pragma unroll
  for (int i4 = 0; i4 < 4; i4++) {
    v8h hv = *(const v8h*)(row + i4 * 8);             // 16B coalesced loads
    #pragma unroll
    for (int e = 0; e < 8; e++) {
      int f = sub * 32 + i4 * 8 + e;
      float h = scale2[f] * (float)hv[e] + shift2[f];
      h = h > 0.f ? h : 0.f;
      d0 += h * w0[f];
      d1 += h * w1[f];
    }
  }
  d0 += __shfl_xor(d0, 1); d0 += __shfl_xor(d0, 2);
  d1 += __shfl_xor(d1, 1); d1 += __shfl_xor(d1, 2);
  if (sub == 0) {
    float s  = tanhf(d0 + bfv[2 * l]);
    float t  = d1 + bfv[2 * l + 1];
    float xo = xcur[2 * samp + xo_col];
    xcur[2 * samp + xo_col] = expf(s) * xo + t;
    sldj[samp] += s;
  }
}

// ---- final sigmoid + log-det correction -----------------------------------
__global__ void k_final(const float* __restrict__ xcur, const float* __restrict__ sldj,
                        float* out) {
  int i = blockIdx.x * blockDim.x + threadIdx.x;
  if (i < BB) {
    float z0 = 1.f / (1.f + expf(-xcur[2 * i]));
    float z1 = 1.f / (1.f + expf(-xcur[2 * i + 1]));
    out[2 * i]     = z0;
    out[2 * i + 1] = z1;
    out[2 * BB + i] = sldj[i] + logf(z0 * (1.f - z0) + FLOW_EPS)
                              + logf(z1 * (1.f - z1) + FLOW_EPS);
  }
}

// ---------------------------------------------------------------------------
extern "C" void kernel_launch(void* const* d_in, const int* in_sizes, int n_in,
                              void* d_out, int out_size, void* d_ws, size_t ws_size,
                              hipStream_t stream) {
  const float* x    = (const float*)d_in[0];
  const float* v1   = (const float*)d_in[1];
  const float* g1   = (const float*)d_in[2];
  // d_in[3] = b1  : cancels under BN1, never read
  const float* bn1g = (const float*)d_in[4];
  const float* bn1b = (const float*)d_in[5];
  const float* v2   = (const float*)d_in[6];
  const float* g2   = (const float*)d_in[7];
  // d_in[8] = b2  : cancels under BN2, never read
  const float* bn2g = (const float*)d_in[9];
  const float* bn2b = (const float*)d_in[10];
  const float* wf   = (const float*)d_in[11];
  const float* bfv  = (const float*)d_in[12];

  // Workspace layout (floats). Requires ~71MB of d_ws.
  float* wsf     = (float*)d_ws;
  float* xcur    = wsf;                       // 2B floats
  float* sldj    = wsf + 2 * BB;              // B floats
  float* ustats  = wsf + 3 * BB;              // 2
  float* h2sum   = ustats + 2;                // 128
  float* h2sumsq = h2sum + HH;                // 128   (ustats..h2sumsq contiguous: 258)
  float* alpha   = wsf + 3 * BB + 258;        // 128
  float* beta    = alpha + HH;                // 128
  float* scale2  = beta + HH;                 // 128
  float* shift2  = scale2 + HH;               // 128
  _Float16* w2h  = (_Float16*)(wsf + 3 * BB + 1024);   // 128*128 f16
  _Float16* H2   = (_Float16*)(wsf + 3 * BB + 9216);   // B*128 f16 (16B aligned)

  k_init<<<BB / 256, 256, 0, stream>>>(x, xcur, sldj);

  for (int l = 0; l < LL; l++) {
    int rev = (l % 2 == 0);
    int xin_col = rev ? 1 : 0;
    int xo_col  = rev ? 0 : 1;

    k_zero<<<2, 256, 0, stream>>>(ustats, 258);
    k_ustat<<<1024, 256, 0, stream>>>(xcur, ustats, xin_col);
    k_prep<<<1, 128, 0, stream>>>(v1, g1, bn1g, bn1b, v2, g2, ustats,
                                  alpha, beta, w2h, l);
    k_gemm<<<GEMM_BLOCKS, 256, 0, stream>>>(xcur, w2h, alpha, beta,
                                            H2, h2sum, h2sumsq, xin_col);
    k_bn2prep<<<1, 128, 0, stream>>>(bn2g, bn2b, h2sum, h2sumsq, scale2, shift2, l);
    k_out<<<(BB * 4) / 256, 256, 0, stream>>>(H2, scale2, shift2, wf, bfv,
                                              xcur, sldj, l, xo_col);
  }

  k_final<<<BB / 256, 256, 0, stream>>>(xcur, sldj, (float*)d_out);
}
